// BertGraphCrossAttention_13297218748517
// MI455X (gfx1250) — matrix operand order repaired and checked by reading, not compiled, not run on hardware
//
#include <hip/hip_runtime.h>
#include <hip/hip_bf16.h>

typedef __bf16 bf16_t;
typedef __attribute__((ext_vector_type(8)))  __bf16 v8bf;
typedef __attribute__((ext_vector_type(16))) __bf16 v16bf;
typedef __attribute__((ext_vector_type(8)))  float  v8f;

#define HN     12
#define DH     64
#define NB     4
#define LQ     512
#define LK     512
#define DMODEL 768

// ---------------- LDS row strides (elements) with bank-conflict padding ----
#define QK_LD 72    // 64 + 8 bf16
#define SC_LD 516   // 512 + 4 floats
#define PB_LD 520   // 512 + 8 bf16
#define VT_LD 520

// Dynamic LDS layout for attention kernel (bytes)
constexpr int OFF_SC   = 0;
constexpr int OFF_PB   = OFF_SC   + 64 * SC_LD * 4;   // 132096
constexpr int OFF_VT   = OFF_PB   + 64 * PB_LD * 2;   // 198656
constexpr int OFF_QL   = OFF_VT   + 64 * VT_LD * 2;   // 265216
constexpr int OFF_KL   = OFF_QL   + 64 * QK_LD * 2;   // 274432
constexpr int OFF_TQC  = OFF_KL   + 64 * QK_LD * 2;   // 283648  float[64][4]
constexpr int OFF_PMAX = OFF_TQC  + 64 * 4 * 4;       // 284672  float[64][4]
constexpr int OFF_PSUM = OFF_PMAX + 64 * 4 * 4;       // 285696  float[64][4]
constexpr int OFF_PCLP = OFF_PSUM + 64 * 4 * 4;       // 286720  float[64][4][4]
constexpr int OFF_PCLS = OFF_PCLP + 64 * 4 * 4 * 4;   // 290816  float[64][4]
constexpr int OFF_RMAX = OFF_PCLS + 64 * 4 * 4;       // 291840  float[64]
constexpr int OFF_RINV = OFF_RMAX + 64 * 4;           // 292096
constexpr int SMEM_ATT = OFF_RINV + 64 * 4;           // 292352 bytes < 320 KB

// Build a v16bf WMMA fragment from two 16-byte-aligned LDS chunks.
__device__ inline v16bf make_frag(const bf16_t* p0, const bf16_t* p1) {
  v8bf a = *(const v8bf*)p0;
  v8bf b = *(const v8bf*)p1;
  v16bf r;
#pragma unroll
  for (int i = 0; i < 8; ++i) { r[i] = a[i]; r[i + 8] = b[i]; }
  return r;
}

// Async global -> LDS copy of 16 bytes per lane (gfx1250 ASYNCcnt path).
// Low 32 bits of a generic LDS pointer are the LDS byte offset (ISA 10.2).
__device__ inline void async_ld_b128(const void* lds_ptr, const void* gptr) {
  const uint32_t lds = (uint32_t)(uintptr_t)lds_ptr;
  const uint64_t ga  = (uint64_t)(uintptr_t)gptr;
  asm volatile("global_load_async_to_lds_b128 %0, %1, off"
               :: "v"(lds), "v"(ga) : "memory");
}
__device__ inline void wait_async0() {
  asm volatile("s_wait_asynccnt 0x0" ::: "memory");
}

// ---------------------------------------------------------------------------
// Kernel B: layernorm of the 3 relation embeddings (keys and values tables)
// ---------------------------------------------------------------------------
__global__ void ln_precompute(const float* __restrict__ dpk, const float* __restrict__ dpv,
                              const float* __restrict__ gk,  const float* __restrict__ bk2,
                              const float* __restrict__ gv,  const float* __restrict__ bv2,
                              float* __restrict__ lnk, float* __restrict__ lnv) {
  __shared__ float sh[6][64];
  __shared__ float mu[6], rs[6];
  const int x = threadIdx.x;           // 0..63  (feature dim)
  const int y = threadIdx.y;           // 0..5   (table*3 + class)
  const int table = y / 3, c = y % 3;
  const float* e = table ? dpv : dpk;
  const float v = e[c * DH + x];
  sh[y][x] = v;
  __syncthreads();
  if (x == 0) {
    float s = 0.f;
    for (int i = 0; i < DH; ++i) s += sh[y][i];
    const float m = s / (float)DH;
    float vs = 0.f;
    for (int i = 0; i < DH; ++i) { const float d = sh[y][i] - m; vs += d * d; }
    mu[y] = m;
    rs[y] = rsqrtf(vs / (float)DH + 1e-5f);
  }
  __syncthreads();
  const float* g  = table ? gv  : gk;
  const float* bb = table ? bv2 : bk2;
  const float o = (v - mu[y]) * rs[y] * g[x] + bb[x];
  (table ? lnv : lnk)[c * DH + x] = o;
}

// ---------------------------------------------------------------------------
// Kernel A: QKV projection GEMM (bf16 WMMA, fp32 accumulate)
//   C[2048 x 768] = X[2048 x 768] @ W[768 x 768] + bias
//   grid: (M/128, N/64, 3) ; block 256 (8 waves) ; out layout [B,H,L,DH] bf16
// ---------------------------------------------------------------------------
__global__ __launch_bounds__(256)
void qkv_gemm(const float* __restrict__ hidden, const float* __restrict__ ctx,
              const float* __restrict__ Wq, const float* __restrict__ bq,
              const float* __restrict__ Wk, const float* __restrict__ bk,
              const float* __restrict__ Wv, const float* __restrict__ bv,
              bf16_t* __restrict__ qbuf, bf16_t* __restrict__ kbuf,
              bf16_t* __restrict__ vbuf) {
  const int z = blockIdx.z;
  const float* X    = (z == 0) ? hidden : ctx;
  const float* W    = (z == 0) ? Wq : (z == 1) ? Wk : Wv;
  const float* bias = (z == 0) ? bq : (z == 1) ? bk : bv;
  bf16_t* dst       = (z == 0) ? qbuf : (z == 1) ? kbuf : vbuf;

  __shared__ bf16_t Al[128][40];   // A tile: 128 rows x 32 k  (+8 pad)
  __shared__ bf16_t Bt[64][40];    // B tile transposed: [n][k]

  const int t    = threadIdx.x;
  const int wv   = t >> 5;
  const int lane = t & 31;
  const int m0   = blockIdx.x * 128;
  const int n0   = blockIdx.y * 64;

  v8f acc[4];
#pragma unroll
  for (int i = 0; i < 4; ++i) acc[i] = (v8f)(0.0f);

  for (int kt = 0; kt < DMODEL / 32; ++kt) {
    const int k0 = kt * 32;
    __syncthreads();
    {   // A tile: 128x32 fp32 -> bf16, 2 threads per row
      const int row = t >> 1, half = t & 1;
      const float* src = X + (size_t)(m0 + row) * DMODEL + k0 + half * 16;
      alignas(16) bf16_t tmp[16];
#pragma unroll
      for (int i = 0; i < 4; ++i) {
        const float4 f = ((const float4*)src)[i];
        tmp[i * 4 + 0] = (bf16_t)f.x; tmp[i * 4 + 1] = (bf16_t)f.y;
        tmp[i * 4 + 2] = (bf16_t)f.z; tmp[i * 4 + 3] = (bf16_t)f.w;
      }
      *(uint4*)&Al[row][half * 16]     = *(const uint4*)&tmp[0];
      *(uint4*)&Al[row][half * 16 + 8] = *(const uint4*)&tmp[8];
    }
    {   // B tile: W[k0..k0+31][n0..n0+63] -> Bt[n][k]
      const int kk = t >> 3, ng = t & 7;
      const float* src = W + (size_t)(k0 + kk) * DMODEL + n0 + ng * 8;
      const float4 f0 = ((const float4*)src)[0];
      const float4 f1 = ((const float4*)src)[1];
      const float v[8] = {f0.x, f0.y, f0.z, f0.w, f1.x, f1.y, f1.z, f1.w};
#pragma unroll
      for (int j = 0; j < 8; ++j) Bt[ng * 8 + j][kk] = (bf16_t)v[j];
    }
    __syncthreads();

    // Batch all fragment loads, then issue the 4 WMMAs back-to-back.
    const int arow = wv * 16 + (lane & 15);
    const int aklo = (lane >> 4) * 8;          // K 0-7/16-23 vs 8-15/24-31
    const v16bf af = make_frag(&Al[arow][aklo], &Al[arow][aklo + 16]);
    const int bks = (lane >> 4) * 8;           // K 0-7/16-23 vs 8-15/24-31
    v16bf bfr[4];
#pragma unroll
    for (int ng = 0; ng < 4; ++ng) {
      const int ncol = ng * 16 + (lane & 15);
      bfr[ng] = make_frag(&Bt[ncol][bks], &Bt[ncol][bks + 16]);
    }
#pragma unroll
    for (int ng = 0; ng < 4; ++ng)
      acc[ng] = __builtin_amdgcn_wmma_f32_16x16x32_bf16(
          false, af, false, bfr[ng], (short)0, acc[ng], false, false);
  }

  // Epilogue: + bias, scatter to [B,H,L,DH] bf16
#pragma unroll
  for (int ng = 0; ng < 4; ++ng) {
    const int n = n0 + ng * 16 + (lane & 15);
    const int h = n >> 6, d = n & 63;
    const float bn = bias[n];
#pragma unroll
    for (int r = 0; r < 8; ++r) {
      const int m  = m0 + wv * 16 + r + ((lane >> 4) ? 8 : 0);
      const int bb = m >> 9, l = m & 511;
      const float val = acc[ng][r] + bn;
      dst[(((size_t)(bb * HN + h)) * LQ + l) * DH + d] = (bf16_t)val;
    }
  }
}

// ---------------------------------------------------------------------------
// Kernel C: attention per (b,h), 64-query tile. grid (LQ/64, B*H), block 256.
// ---------------------------------------------------------------------------
__global__ __launch_bounds__(256)
void attention(const bf16_t* __restrict__ qbuf, const bf16_t* __restrict__ kbuf,
               const bf16_t* __restrict__ vbuf, const int* __restrict__ arc,
               const float* __restrict__ mask, const float* __restrict__ lnk,
               const float* __restrict__ lnv, float* __restrict__ out) {
  extern __shared__ char smem[];
  float*  scoreS = (float*)(smem + OFF_SC);    // [64][SC_LD] fp32 scores
  bf16_t* pbf    = (bf16_t*)(smem + OFF_PB);   // [64][PB_LD] bf16 probs
  bf16_t* vt     = (bf16_t*)(smem + OFF_VT);   // [64][VT_LD] V transposed [d][k]
  bf16_t* qls    = (bf16_t*)(smem + OFF_QL);   // [64][QK_LD]
  bf16_t* kls    = (bf16_t*)(smem + OFF_KL);   // [64][QK_LD]
  float*  tqc    = (float*)(smem + OFF_TQC);   // [64][4] q . ln_dpk[c]
  float*  pmax   = (float*)(smem + OFF_PMAX);
  float*  psum   = (float*)(smem + OFF_PSUM);
  float*  pclsP  = (float*)(smem + OFF_PCLP);
  float*  pcls   = (float*)(smem + OFF_PCLS);  // [64][4] class prob mass
  float*  rowmax = (float*)(smem + OFF_RMAX);
  float*  rowinv = (float*)(smem + OFF_RINV);

  const int t    = threadIdx.x;
  const int wv   = t >> 5;
  const int lane = t & 31;
  const int q0   = blockIdx.x * 64;
  const int bh   = blockIdx.y;
  const int b    = bh / HN, h = bh % HN;
  const size_t headoff = (size_t)bh * LQ * DH;

  // ---- Phase 0: async-copy 64x64 Q tile to LDS (2 x b128 per lane) ----
  {
    const int row = t >> 2, cs = (t & 3) * 16;
    const bf16_t* g = qbuf + headoff + (size_t)(q0 + row) * DH + cs;
    bf16_t* l = &qls[row * QK_LD + cs];
    async_ld_b128(l, g);
    async_ld_b128(l + 8, g + 8);
    wait_async0();
  }
  __syncthreads();

  // ---- Phase 1: t_c = q . ln_dpk[c], 3 classes x 64 rows ----
  if (t < 192) {
    const int r = t & 63, c = t >> 6;
    const float* lk = lnk + c * DH;
    float s = 0.f;
    for (int d = 0; d < DH; ++d) s += (float)qls[r * QK_LD + d] * lk[d];
    tqc[r * 4 + c] = s;
  }
  __syncthreads();

  // ---- Phase 2: scores = QK^T/8 + t[arc] + mask, tiled over k ----
  for (int kt = 0; kt < LK / 64; ++kt) {
    const int kb = kt * 64;
    {   // K tile via async global->LDS copy
      const int row = t >> 2, cs = (t & 3) * 16;
      const bf16_t* g = kbuf + headoff + (size_t)(kb + row) * DH + cs;
      bf16_t* l = &kls[row * QK_LD + cs];
      async_ld_b128(l, g);
      async_ld_b128(l + 8, g + 8);
    }
    {   // V tile, transposed into vt[d][kb+kk] (overlaps the async K copy)
      const int kk = t >> 2, ds0 = (t & 3) * 16;
      alignas(16) bf16_t vu[16];
      const uint4* src = (const uint4*)(vbuf + headoff + (size_t)(kb + kk) * DH + ds0);
      *(uint4*)&vu[0] = src[0];
      *(uint4*)&vu[8] = src[1];
#pragma unroll
      for (int j = 0; j < 16; ++j) vt[(ds0 + j) * VT_LD + kb + kk] = vu[j];
    }
    wait_async0();
    __syncthreads();

    const int qg = (wv & 3) * 16;        // 16 query rows per wave
    const int kg = (wv >> 2) * 32;       // 32 key cols per wave (2 tiles)
    v8f acc0 = (v8f)(0.0f), acc1 = (v8f)(0.0f);
#pragma unroll
    for (int dstep = 0; dstep < DH; dstep += 32) {
      const int arow = qg + (lane & 15);
      const int aklo = dstep + (lane >> 4) * 8;
      const int bks  = dstep + (lane >> 4) * 8;
      const int kc0  = kg + (lane & 15);
      const int kc1  = kg + 16 + (lane & 15);
      const v16bf af  = make_frag(&qls[arow * QK_LD + aklo],
                                  &qls[arow * QK_LD + aklo + 16]);
      const v16bf bf0 = make_frag(&kls[kc0 * QK_LD + bks], &kls[kc0 * QK_LD + bks + 16]);
      const v16bf bf1 = make_frag(&kls[kc1 * QK_LD + bks], &kls[kc1 * QK_LD + bks + 16]);
      acc0 = __builtin_amdgcn_wmma_f32_16x16x32_bf16(false, af, false, bf0,
                                                     (short)0, acc0, false, false);
      acc1 = __builtin_amdgcn_wmma_f32_16x16x32_bf16(false, af, false, bf1,
                                                     (short)0, acc1, false, false);
    }
    const int qrb = qg + ((lane >> 4) ? 8 : 0);
#pragma unroll
    for (int s = 0; s < 2; ++s) {
      const v8f a = s ? acc1 : acc0;
      const int kgl = kb + kg + s * 16 + (lane & 15);
      const float mk = mask[b * LK + kgl];
#pragma unroll
      for (int r = 0; r < 8; ++r) {
        const int qr  = qrb + r;
        const int cls = arc[((size_t)(b * LQ) + q0 + qr) * LK + kgl];
        scoreS[qr * SC_LD + kgl] = a[r] * 0.125f + tqc[qr * 4 + cls] + mk;
      }
    }
    __syncthreads();
  }

  // ---- Phase 3: softmax + per-class probability masses ----
  {
    const int r = t >> 2, p = t & 3;
    float* row = scoreS + r * SC_LD;
    const int c0 = p * 128;
    float m = -1e30f;
    for (int c = 0; c < 128; ++c) m = fmaxf(m, row[c0 + c]);
    pmax[r * 4 + p] = m;
    __syncthreads();
    if (p == 0)
      rowmax[r] = fmaxf(fmaxf(pmax[r * 4], pmax[r * 4 + 1]),
                        fmaxf(pmax[r * 4 + 2], pmax[r * 4 + 3]));
    __syncthreads();
    const float mm = rowmax[r];
    float sum = 0.f, cl0 = 0.f, cl1 = 0.f, cl2 = 0.f;
    const int* arow = arc + ((size_t)(b * LQ) + q0 + r) * LK;
    for (int c = 0; c < 128; ++c) {
      const float e = __expf(row[c0 + c] - mm);
      row[c0 + c] = e;
      sum += e;
      const int cls = arow[c0 + c];
      cl0 += (cls == 0) ? e : 0.f;
      cl1 += (cls == 1) ? e : 0.f;
      cl2 += (cls == 2) ? e : 0.f;
    }
    psum[r * 4 + p] = sum;
    pclsP[(r * 4 + p) * 4 + 0] = cl0;
    pclsP[(r * 4 + p) * 4 + 1] = cl1;
    pclsP[(r * 4 + p) * 4 + 2] = cl2;
    __syncthreads();
    if (p == 0) {
      const float s = psum[r * 4] + psum[r * 4 + 1] + psum[r * 4 + 2] + psum[r * 4 + 3];
      const float inv = 1.f / s;
      rowinv[r] = inv;
      for (int c = 0; c < 3; ++c) {
        const float cc = pclsP[(r * 4 + 0) * 4 + c] + pclsP[(r * 4 + 1) * 4 + c] +
                         pclsP[(r * 4 + 2) * 4 + c] + pclsP[(r * 4 + 3) * 4 + c];
        pcls[r * 4 + c] = cc * inv;
      }
    }
    __syncthreads();
    const float inv = rowinv[r];
    for (int c = 0; c < 128; ++c)
      pbf[r * PB_LD + c0 + c] = (bf16_t)(row[c0 + c] * inv);
  }
  __syncthreads();

  // ---- Phase 4: ctx = P @ V via WMMA over full Lk ----
  const int qg2 = (wv & 3) * 16;
  const int dg2 = (wv >> 2) * 32;
  v8f o0 = (v8f)(0.0f), o1 = (v8f)(0.0f);
  for (int ks = 0; ks < LK; ks += 32) {
    const int arow = qg2 + (lane & 15);
    const int aklo = ks + (lane >> 4) * 8;
    const int bks  = ks + (lane >> 4) * 8;
    const int dc0  = dg2 + (lane & 15);
    const int dc1  = dg2 + 16 + (lane & 15);
    const v16bf af  = make_frag(&pbf[arow * PB_LD + aklo],
                                &pbf[arow * PB_LD + aklo + 16]);
    const v16bf bf0 = make_frag(&vt[dc0 * VT_LD + bks], &vt[dc0 * VT_LD + bks + 16]);
    const v16bf bf1 = make_frag(&vt[dc1 * VT_LD + bks], &vt[dc1 * VT_LD + bks + 16]);
    o0 = __builtin_amdgcn_wmma_f32_16x16x32_bf16(false, af, false, bf0,
                                                 (short)0, o0, false, false);
    o1 = __builtin_amdgcn_wmma_f32_16x16x32_bf16(false, af, false, bf1,
                                                 (short)0, o1, false, false);
  }

  // ---- Phase 5: + sum_c pcls_c * ln_dpv[c], store fp32 [B, Lq, D] ----
  const int qrb = qg2 + ((lane >> 4) ? 8 : 0);
#pragma unroll
  for (int s = 0; s < 2; ++s) {
    const v8f a = s ? o1 : o0;
    const int dc = dg2 + s * 16 + (lane & 15);
#pragma unroll
    for (int r = 0; r < 8; ++r) {
      const int qr = qrb + r;
      const float add = pcls[qr * 4 + 0] * lnv[0 * DH + dc] +
                        pcls[qr * 4 + 1] * lnv[1 * DH + dc] +
                        pcls[qr * 4 + 2] * lnv[2 * DH + dc];
      out[((size_t)(b * LQ) + q0 + qr) * DMODEL + h * DH + dc] = a[r] + add;
    }
  }
}

// ---------------------------------------------------------------------------
extern "C" void kernel_launch(void* const* d_in, const int* in_sizes, int n_in,
                              void* d_out, int out_size, void* d_ws, size_t ws_size,
                              hipStream_t stream) {
  (void)in_sizes; (void)n_in; (void)out_size; (void)ws_size;
  const float* hidden = (const float*)d_in[0];
  const float* contex = (const float*)d_in[1];
  const float* mask   = (const float*)d_in[2];
  const int*   arc    = (const int*)d_in[3];
  const float* Wq = (const float*)d_in[4];
  const float* bq = (const float*)d_in[5];
  const float* Wk = (const float*)d_in[6];
  const float* bk = (const float*)d_in[7];
  const float* Wv = (const float*)d_in[8];
  const float* bv = (const float*)d_in[9];
  const float* dpk  = (const float*)d_in[10];
  const float* dpv  = (const float*)d_in[11];
  const float* lnkg = (const float*)d_in[12];
  const float* lnkb = (const float*)d_in[13];
  const float* lnvg = (const float*)d_in[14];
  const float* lnvb = (const float*)d_in[15];
  float* out = (float*)d_out;

  const size_t nper = (size_t)NB * HN * LQ * DH;     // 1,572,864 elems
  bf16_t* qbuf = (bf16_t*)d_ws;
  bf16_t* kbuf = qbuf + nper;
  bf16_t* vbuf = kbuf + nper;
  float*  lnk  = (float*)(vbuf + nper);
  float*  lnv  = lnk + 3 * DH;

  ln_precompute<<<dim3(1), dim3(64, 6), 0, stream>>>(dpk, dpv, lnkg, lnkb,
                                                     lnvg, lnvb, lnk, lnv);
  qkv_gemm<<<dim3(16, 12, 3), dim3(256), 0, stream>>>(hidden, contex, Wq, bq,
                                                      Wk, bk, Wv, bv,
                                                      qbuf, kbuf, vbuf);
  attention<<<dim3(LQ / 64, NB * HN), dim3(256), SMEM_ATT, stream>>>(
      qbuf, kbuf, vbuf, arc, mask, lnk, lnv, out);
}
